// MentionScorer_3470333575435
// MI455X (gfx1250) — matrix operand order repaired and checked by reading, not compiled, not run on hardware
//
#include <hip/hip_runtime.h>

// Sizes (fixed by the reference)
#define T_TOK 4096
#define H_DIM 1024
#define E_DIM 512
#define N_SPAN 40960
#define L_MAX 10
#define DE_DIM 20
#define AH_DIM 1024
#define SH_DIM 1024
#define DS_DIM 2580          // 2*H + E + DE
#define DS_PAD 2592          // next multiple of 32 (zero-padded K)
#define CHUNK_K 864          // 2592 / 3, multiple of 32
#define CHUNK_P 872          // padded LDS row stride (halves) for bank spread

typedef __attribute__((ext_vector_type(16))) _Float16 v16h;
typedef __attribute__((ext_vector_type(8)))  _Float16 v8h;
typedef __attribute__((ext_vector_type(8)))  float    v8f;

__device__ __forceinline__ v16h cat8(v8h a, v8h b) {
  return __builtin_shufflevector(a, b, 0,1,2,3,4,5,6,7,8,9,10,11,12,13,14,15);
}

// ---------------------------------------------------------------------------
// Pre-pass: transpose + fp32->fp16 convert weight matrices so the WMMA
// B-fragment (lane<16: N=lane,K=0..7,16..23 ; lane>=16: N=lane-16,K=8..15,24..31) is
// two 8-half (b128) loads per lane.
// ---------------------------------------------------------------------------
__global__ void k_trans_w1(const float* __restrict__ W1, _Float16* __restrict__ W1t) {
  int i = blockIdx.x * 256 + threadIdx.x;      // over AH*H
  int n = i >> 10;                             // output row (hidden unit)
  int k = i & (H_DIM - 1);                     // output col (input dim)
  W1t[i] = (_Float16)W1[(size_t)k * AH_DIM + n];
}

__global__ void k_trans_w3(const float* __restrict__ W3, _Float16* __restrict__ W3t) {
  int i = blockIdx.x * 256 + threadIdx.x;      // over SH*DS_PAD
  int n = i / DS_PAD;
  int k = i - n * DS_PAD;
  W3t[i] = (k < DS_DIM) ? (_Float16)W3[(size_t)k * SH_DIM + n] : (_Float16)0.0f;
}

// ---------------------------------------------------------------------------
// Kernel 1: attn[t] = relu(states[t] @ W1 + b1) @ W2 + b2     (per token)
// One block = one 16-row tile; A staged to LDS as f16; 8 waves x 8 N-tiles.
// ---------------------------------------------------------------------------
__global__ void __launch_bounds__(256) k_attn(
    const float* __restrict__ states, const _Float16* __restrict__ W1t,
    const float* __restrict__ b1, const float* __restrict__ W2,
    const float* __restrict__ b2, float* __restrict__ attn) {
  __shared__ __align__(16) _Float16 Ash[16 * 1032];   // 16 x 1024, padded stride
  __shared__ float ssc[16];
  const int tid  = threadIdx.x;
  const int row0 = blockIdx.x * 16;
  if (tid < 16) ssc[tid] = 0.0f;
  for (int i = tid; i < 16 * H_DIM; i += 256) {
    int r = i >> 10, c = i & (H_DIM - 1);
    Ash[r * 1032 + c] = (_Float16)states[(size_t)(row0 + r) * H_DIM + c];
  }
  __syncthreads();

  const int lane = tid & 31, wave = tid >> 5;
  const int col  = lane & 15;
  const int hi   = (lane >> 4) & 1;
  const _Float16* arow = Ash + col * 1032 + hi * 8;

  float sacc[8] = {0.f, 0.f, 0.f, 0.f, 0.f, 0.f, 0.f, 0.f};
  for (int tt = 0; tt < 8; ++tt) {
    const int n0 = (wave * 8 + tt) * 16;
    const _Float16* brow = W1t + (size_t)(n0 + col) * H_DIM + hi * 8;
    v8f c = {};
    for (int k = 0; k < H_DIM; k += 32) {
      v8h a0 = *(const v8h*)(arow + k);
      v8h a1 = *(const v8h*)(arow + k + 16);
      v8h b0 = *(const v8h*)(brow + k);
      v8h b1h = *(const v8h*)(brow + k + 16);
      c = __builtin_amdgcn_wmma_f32_16x16x32_f16(
              false, cat8(a0, a1), false, cat8(b0, b1h), (short)0, c, false, false);
    }
    const float bias = b1[n0 + col];
    const float w2v  = W2[n0 + col];
    #pragma unroll
    for (int r = 0; r < 8; ++r) {
      float h = c[r] + bias;
      h = h > 0.f ? h : 0.f;
      sacc[r] = fmaf(h, w2v, sacc[r]);
    }
  }
  #pragma unroll
  for (int r = 0; r < 8; ++r) {
    sacc[r] += __shfl_xor(sacc[r], 1, 32);
    sacc[r] += __shfl_xor(sacc[r], 2, 32);
    sacc[r] += __shfl_xor(sacc[r], 4, 32);
    sacc[r] += __shfl_xor(sacc[r], 8, 32);
  }
  if ((lane & 15) == 0) {
    #pragma unroll
    for (int r = 0; r < 8; ++r) atomicAdd(&ssc[hi * 8 + r], sacc[r]);
  }
  __syncthreads();
  if (tid < 16) attn[row0 + tid] = ssc[tid] + b2[0];
}

// ---------------------------------------------------------------------------
// Kernel 2: per-span masked softmax over attn logits, weighted embed sum,
// gather endpoints + length embedding -> states_avg rows (written to d_out).
// One wave per span (span params are wave-uniform).
// ---------------------------------------------------------------------------
__global__ void __launch_bounds__(256) k_assemble(
    const float* __restrict__ states, const float* __restrict__ embeds,
    const int* __restrict__ starts, const int* __restrict__ lens,
    const float* __restrict__ table, const float* __restrict__ attn,
    float* __restrict__ savg) {
  const int lane = threadIdx.x & 31;
  const int n    = blockIdx.x * 8 + (threadIdx.x >> 5);
  const int start = starts[n];
  const int len   = lens[n];
  const int end   = start + len - 1;

  float wv[L_MAX];
  float m = -3.4e38f;
  #pragma unroll
  for (int j = 0; j < L_MAX; ++j)
    if (j < len) { float a = attn[start + j]; wv[j] = a; m = fmaxf(m, a); }
  float s = 0.f;
  #pragma unroll
  for (int j = 0; j < L_MAX; ++j)
    if (j < len) { float e = __expf(wv[j] - m); wv[j] = e; s += e; }
  const float inv = 1.0f / s;

  float* orow = savg + (size_t)n * DS_DIM;
  const float4* s0 = (const float4*)(states + (size_t)start * H_DIM);
  const float4* s1 = (const float4*)(states + (size_t)end * H_DIM);
  float4* o = (float4*)orow;
  #pragma unroll
  for (int i = 0; i < 8; ++i) o[lane + 32 * i] = s0[lane + 32 * i];
  #pragma unroll
  for (int i = 0; i < 8; ++i) o[256 + lane + 32 * i] = s1[lane + 32 * i];

  float4 acc[4];
  #pragma unroll
  for (int i = 0; i < 4; ++i) acc[i] = make_float4(0.f, 0.f, 0.f, 0.f);
  #pragma unroll
  for (int j = 0; j < L_MAX; ++j) {
    if (j < len) {
      const float4* er = (const float4*)(embeds + (size_t)(start + j) * E_DIM);
      const float wj = wv[j] * inv;
      #pragma unroll
      for (int i = 0; i < 4; ++i) {
        float4 v = er[lane + 32 * i];
        acc[i].x = fmaf(wj, v.x, acc[i].x);
        acc[i].y = fmaf(wj, v.y, acc[i].y);
        acc[i].z = fmaf(wj, v.z, acc[i].z);
        acc[i].w = fmaf(wj, v.w, acc[i].w);
      }
    }
  }
  #pragma unroll
  for (int i = 0; i < 4; ++i) o[512 + lane + 32 * i] = acc[i];
  if (lane < DE_DIM) orow[2 * H_DIM + E_DIM + lane] = table[len * DE_DIM + lane];
}

// ---------------------------------------------------------------------------
// Kernel 3: scores = relu(states_avg @ W3 + b3) @ W4 + b4   (the big GEMM)
// M=32 rows per block (two 16-row tiles per wave) so each B fragment feeds
// TWO WMMAs -> halves B-operand L2 traffic vs 16-row blocking.
// K = 2592 (padded) in 3 LDS chunks of 864.
// ---------------------------------------------------------------------------
__global__ void __launch_bounds__(256) k_score(
    const float* __restrict__ savg, const _Float16* __restrict__ W3t,
    const float* __restrict__ b3, const float* __restrict__ W4,
    const float* __restrict__ b4, float* __restrict__ scores) {
  __shared__ __align__(16) _Float16 Ash[32 * CHUNK_P];  // ~55.8 KB
  __shared__ float ssc[32];
  const int tid  = threadIdx.x;
  const int row0 = blockIdx.x * 32;
  if (tid < 32) ssc[tid] = 0.0f;

  const int lane = tid & 31, wave = tid >> 5;
  const int col  = lane & 15;
  const int hi   = (lane >> 4) & 1;
  const _Float16* arow0 = Ash + col * CHUNK_P + hi * 8;            // rows 0..15
  const _Float16* arow1 = arow0 + 16 * CHUNK_P;                    // rows 16..31

  v8f cacc0[8] = {};   // accumulators for row-tile 0, 8 N-tiles
  v8f cacc1[8] = {};   // accumulators for row-tile 1, 8 N-tiles
  for (int ch = 0; ch < 3; ++ch) {
    const int kb = ch * CHUNK_K;
    __syncthreads();   // protect previous chunk's reads
    for (int i = tid; i < 32 * CHUNK_K; i += 256) {
      int r = i / CHUNK_K, cc = i - r * CHUNK_K;
      int k = kb + cc;
      float v = (k < DS_DIM) ? savg[(size_t)(row0 + r) * DS_DIM + k] : 0.f;
      Ash[r * CHUNK_P + cc] = (_Float16)v;
    }
    __syncthreads();
    #pragma unroll
    for (int tt = 0; tt < 8; ++tt) {
      const int n0 = (wave * 8 + tt) * 16;
      const _Float16* brow = W3t + (size_t)(n0 + col) * DS_PAD + kb + hi * 8;
      v8f c0 = cacc0[tt];
      v8f c1 = cacc1[tt];
      for (int k = 0; k < CHUNK_K; k += 32) {
        v8h b0 = *(const v8h*)(brow + k);
        v8h b1h = *(const v8h*)(brow + k + 16);
        v16h b = cat8(b0, b1h);
        v8h a00 = *(const v8h*)(arow0 + k);
        v8h a01 = *(const v8h*)(arow0 + k + 16);
        c0 = __builtin_amdgcn_wmma_f32_16x16x32_f16(
                 false, cat8(a00, a01), false, b, (short)0, c0, false, false);
        v8h a10 = *(const v8h*)(arow1 + k);
        v8h a11 = *(const v8h*)(arow1 + k + 16);
        c1 = __builtin_amdgcn_wmma_f32_16x16x32_f16(
                 false, cat8(a10, a11), false, b, (short)0, c1, false, false);
      }
      cacc0[tt] = c0;
      cacc1[tt] = c1;
    }
  }

  float sacc0[8] = {0.f, 0.f, 0.f, 0.f, 0.f, 0.f, 0.f, 0.f};
  float sacc1[8] = {0.f, 0.f, 0.f, 0.f, 0.f, 0.f, 0.f, 0.f};
  #pragma unroll
  for (int tt = 0; tt < 8; ++tt) {
    const int n0 = (wave * 8 + tt) * 16;
    const float bias = b3[n0 + col];
    const float w4v  = W4[n0 + col];
    v8f c0 = cacc0[tt];
    v8f c1 = cacc1[tt];
    #pragma unroll
    for (int r = 0; r < 8; ++r) {
      float h0 = c0[r] + bias;
      h0 = h0 > 0.f ? h0 : 0.f;
      sacc0[r] = fmaf(h0, w4v, sacc0[r]);
      float h1 = c1[r] + bias;
      h1 = h1 > 0.f ? h1 : 0.f;
      sacc1[r] = fmaf(h1, w4v, sacc1[r]);
    }
  }
  #pragma unroll
  for (int r = 0; r < 8; ++r) {
    sacc0[r] += __shfl_xor(sacc0[r], 1, 32);
    sacc0[r] += __shfl_xor(sacc0[r], 2, 32);
    sacc0[r] += __shfl_xor(sacc0[r], 4, 32);
    sacc0[r] += __shfl_xor(sacc0[r], 8, 32);
    sacc1[r] += __shfl_xor(sacc1[r], 1, 32);
    sacc1[r] += __shfl_xor(sacc1[r], 2, 32);
    sacc1[r] += __shfl_xor(sacc1[r], 4, 32);
    sacc1[r] += __shfl_xor(sacc1[r], 8, 32);
  }
  if ((lane & 15) == 0) {
    #pragma unroll
    for (int r = 0; r < 8; ++r) {
      atomicAdd(&ssc[hi * 8 + r], sacc0[r]);
      atomicAdd(&ssc[16 + hi * 8 + r], sacc1[r]);
    }
  }
  __syncthreads();
  if (tid < 32) scores[row0 + tid] = ssc[tid] + b4[0];
}

// ---------------------------------------------------------------------------
extern "C" void kernel_launch(void* const* d_in, const int* in_sizes, int n_in,
                              void* d_out, int out_size, void* d_ws, size_t ws_size,
                              hipStream_t stream) {
  const float* states = (const float*)d_in[0];
  const float* embeds = (const float*)d_in[1];
  const int*   starts = (const int*)d_in[2];
  const int*   lens   = (const int*)d_in[3];
  const float* W1     = (const float*)d_in[4];
  const float* b1     = (const float*)d_in[5];
  const float* W2     = (const float*)d_in[6];
  const float* b2     = (const float*)d_in[7];
  const float* table  = (const float*)d_in[8];
  const float* W3     = (const float*)d_in[9];
  const float* b3     = (const float*)d_in[10];
  const float* W4     = (const float*)d_in[11];
  const float* b4     = (const float*)d_in[12];

  // Outputs: scores [N] then states_avg [N, DS] (reference tuple order).
  float* scores = (float*)d_out;
  float* savg   = scores + N_SPAN;

  // Workspace layout (~7.5 MB): attn[T] f32, W1t f16 [AH][H], W3t f16 [SH][DS_PAD]
  char* ws = (char*)d_ws;
  float*    attn = (float*)ws;                                   //  16 KB
  _Float16* W1t  = (_Float16*)(ws + 16384);                      //   2 MB
  _Float16* W3t  = (_Float16*)(ws + 16384 + (size_t)AH_DIM * H_DIM * 2);

  k_trans_w1<<<(AH_DIM * H_DIM) / 256, 256, 0, stream>>>(W1, W1t);
  k_trans_w3<<<(SH_DIM * DS_PAD) / 256, 256, 0, stream>>>(W3, W3t);
  k_attn<<<T_TOK / 16, 256, 0, stream>>>(states, W1t, b1, W2, b2, attn);
  k_assemble<<<N_SPAN / 8, 256, 0, stream>>>(states, embeds, starts, lens,
                                             table, attn, savg);
  k_score<<<N_SPAN / 32, 256, 0, stream>>>(savg, W3t, b3, W4, b4, scores);
}
